// PixelSNAIL_78786880078141
// MI455X (gfx1250) — matrix operand order repaired and checked by reading, not compiled, not run on hardware
//
#include <hip/hip_runtime.h>
#include <hip/hip_bf16.h>

typedef __bf16 bf16_t;
typedef __attribute__((ext_vector_type(16))) __bf16       v16bf;
typedef __attribute__((ext_vector_type(8)))  float        v8f;
typedef __attribute__((ext_vector_type(4)))  unsigned int u32x4;

#define B_     8
#define S_     1024
#define CDIM   256
#define EMB    512
#define HEADS_ 8
#define HDIM   64

__device__ __forceinline__ float fast_exp(float x) {
#if __has_builtin(__builtin_amdgcn_exp2f)
  return __builtin_amdgcn_exp2f(x * 1.4426950408889634f);
#else
  return exp2f(x * 1.4426950408889634f);
#endif
}

// Assemble a v16bf fragment from two 16-byte chunks (elements 0..7 and 8..15).
__device__ __forceinline__ v16bf load_frag(const bf16_t* p0, const bf16_t* p1) {
  union { v16bf v; u32x4 u[2]; } f;
  f.u[0] = *(const u32x4*)p0;
  f.u[1] = *(const u32x4*)p1;
  return f.v;
}

__device__ __forceinline__ v8f wmma_bf16(v16bf a, v16bf b, v8f c) {
  return __builtin_amdgcn_wmma_f32_16x16x32_bf16(false, a, false, b, (short)0, c,
                                                 false, false);
}

// ---------------------------------------------------------------------------
// 1) f32 -> bf16 weight conversion (Wq, Wk, Wv are (E, C) row-major, kept as-is)
// ---------------------------------------------------------------------------
__global__ void cvt_weights(const float* __restrict__ wq, const float* __restrict__ wk,
                            const float* __restrict__ wv,
                            bf16_t* __restrict__ oq, bf16_t* __restrict__ ok,
                            bf16_t* __restrict__ ov) {
  int i = blockIdx.x * blockDim.x + threadIdx.x;
  const int per = EMB * CDIM;
  if (i >= 3 * per) return;
  int sel = i / per;
  int r = i - sel * per;
  const float* s = (sel == 0) ? wq : (sel == 1 ? wk : wv);
  bf16_t* o = (sel == 0) ? oq : (sel == 1 ? ok : ov);
  o[r] = (bf16_t)s[r];
}

// ---------------------------------------------------------------------------
// 2) input transpose+convert: (B, C, S) f32 -> (B, S, C) bf16 for query & key
// ---------------------------------------------------------------------------
__global__ void cvt_inputs(const float* __restrict__ q, const float* __restrict__ k,
                           bf16_t* __restrict__ xq, bf16_t* __restrict__ xk) {
  int i = blockIdx.x * blockDim.x + threadIdx.x;
  if (i >= 2 * B_ * S_ * CDIM) return;
  int c = i & (CDIM - 1);
  int s = (i >> 8) & (S_ - 1);
  int b = (i >> 18) & 7;
  int sel = i >> 21;
  const float* src = sel ? k : q;
  bf16_t* dst = sel ? xk : xq;
  dst[((size_t)b * S_ + s) * CDIM + c] = (bf16_t)src[((size_t)b * CDIM + c) * S_ + s];
}

// ---------------------------------------------------------------------------
// 3) projections: Out[s,e] = sum_c X[s,c] * W[e,c] + bias[e]
//    which=0 -> Q (B,H,S,D) ; which=1 -> K (B,H,S,D) ; which=2 -> V^T (B,H,D,S)
//    grid: (S/16, EMB/128, 3*B), block 256 (8 waves; wave owns one 16-wide e tile)
// ---------------------------------------------------------------------------
__global__ __launch_bounds__(256) void proj_kernel(
    const bf16_t* __restrict__ Xq, const bf16_t* __restrict__ Xk,
    const bf16_t* __restrict__ Wq, const bf16_t* __restrict__ Wk,
    const bf16_t* __restrict__ Wv,
    const float* __restrict__ bq, const float* __restrict__ bk,
    const float* __restrict__ bv,
    bf16_t* __restrict__ Qo, bf16_t* __restrict__ Ko, bf16_t* __restrict__ Vto) {
  const int wave = threadIdx.x >> 5;
  const int lane = threadIdx.x & 31;
  const int ln = lane & 15;
  const int hi = lane >> 4;
  const int which = blockIdx.z >> 3;
  const int b = blockIdx.z & 7;
  const int sbase = blockIdx.x * 16;
  const int ebase = blockIdx.y * 128 + wave * 16;

  const bf16_t* X = (which == 0) ? Xq : Xk;
  const bf16_t* W = (which == 0) ? Wq : (which == 1 ? Wk : Wv);
  const float* bias = (which == 0) ? bq : (which == 1 ? bk : bv);

  // A: X tile, M = s row (lane%16), K = c (contiguous 16 per half-wave: c 0-15 / 16-31)
  const bf16_t* xr = X + ((size_t)b * S_ + sbase + ln) * CDIM;
  // B: W^T tile, N = e (lane%16), K-dim rows 0-15 / 16-31 per half (contiguous c)
  const bf16_t* wr = W + (size_t)(ebase + ln) * CDIM;

  v8f acc = {};
#pragma unroll
  for (int cb = 0; cb < CDIM; cb += 32) {
    v16bf a = load_frag(xr + cb + hi * 16, xr + cb + hi * 16 + 8);
    v16bf w = load_frag(wr + cb + hi * 16, wr + cb + hi * 16 + 8);
    acc = wmma_bf16(a, w, acc);
  }

  const int e = ebase + ln;
  const float bb = bias[e];
  const int h = e >> 6;
  const int d = e & 63;
#pragma unroll
  for (int v = 0; v < 8; ++v) {
    const int srow = sbase + v + 8 * hi;
    const float r = acc[v] + bb;
    if (which == 0)
      Qo[(((size_t)b * HEADS_ + h) * S_ + srow) * HDIM + d] = (bf16_t)r;
    else if (which == 1)
      Ko[(((size_t)b * HEADS_ + h) * S_ + srow) * HDIM + d] = (bf16_t)r;
    else
      Vto[(((size_t)b * HEADS_ + h) * HDIM + d) * S_ + srow] = (bf16_t)r;
  }
}

// ---------------------------------------------------------------------------
// 4) causal flash attention.
//    grid: (S/128, B*H), block 256 = 8 waves; wave owns a 16-row q tile.
//    Streams k in chunks of 32 with online softmax; P transposed C->A layout
//    through per-wave private LDS (same-wave LDS is in-order; no barrier).
// ---------------------------------------------------------------------------
__global__ __launch_bounds__(256) void attn_kernel(
    const bf16_t* __restrict__ Q, const bf16_t* __restrict__ K,
    const bf16_t* __restrict__ Vt, float* __restrict__ out) {
  __shared__ alignas(16) bf16_t pbuf[8][16 * 32];

  const int wave = threadIdx.x >> 5;
  const int lane = threadIdx.x & 31;
  const int ln = lane & 15;
  const int hi = lane >> 4;
  const int bh = blockIdx.y;
  const int b = bh >> 3;
  const int h = bh & 7;
  const int qbase = blockIdx.x * 128 + wave * 16;

  const bf16_t* Qp = Q + (size_t)bh * S_ * HDIM;
  const bf16_t* Kp = K + (size_t)bh * S_ * HDIM;
  const bf16_t* Vp = Vt + (size_t)bh * HDIM * S_;

  // Q A-fragments for d 0..31 and 32..63
  const bf16_t* qr = Qp + (size_t)(qbase + ln) * HDIM;
  const v16bf aq0 = load_frag(qr + hi * 16, qr + hi * 16 + 8);
  const v16bf aq1 = load_frag(qr + 32 + hi * 16, qr + 40 + hi * 16);

  v8f acc0 = {}, acc1 = {}, acc2 = {}, acc3 = {};
  float mrow[8], ssum[8];
#pragma unroll
  for (int v = 0; v < 8; ++v) { mrow[v] = -3.0e38f; ssum[v] = 0.0f; }

  bf16_t* pb = pbuf[wave];

  for (int kb = 0; kb < qbase + 16; kb += 32) {
    // K B-fragments: B[d][k_idx] = K[k_idx, d]; per lane 32B contiguous in d.
    const bf16_t* k0 = Kp + (size_t)(kb + ln) * HDIM + hi * 16;
    const bf16_t* k1 = Kp + (size_t)(kb + 16 + ln) * HDIM + hi * 16;
    const v16bf bk00 = load_frag(k0, k0 + 8);
    const v16bf bk01 = load_frag(k0 + 32, k0 + 40);
    const v16bf bk10 = load_frag(k1, k1 + 8);
    const v16bf bk11 = load_frag(k1 + 32, k1 + 40);

    if (kb + 32 < qbase + 16)
      __builtin_prefetch(Kp + (size_t)(kb + 32 + ln) * HDIM, 0, 1);

    v8f s0 = {}, s1 = {};
    s0 = wmma_bf16(aq0, bk00, s0);
    s0 = wmma_bf16(aq1, bk01, s0);
    s1 = wmma_bf16(aq0, bk10, s1);
    s1 = wmma_bf16(aq1, bk11, s1);

    // scale + causal mask (C layout: row M = v + 8*hi, col N = ln)
    float p0[8], p1[8], t[8];
#pragma unroll
    for (int v = 0; v < 8; ++v) {
      const int qrow = qbase + v + 8 * hi;
      float x0 = s0[v] * 0.125f;
      float x1 = s1[v] * 0.125f;
      if (kb + ln > qrow) x0 = -1.0e30f;
      if (kb + 16 + ln > qrow) x1 = -1.0e30f;
      p0[v] = x0;
      p1[v] = x1;
      t[v] = fmaxf(x0, x1);
    }
    // row max across the 16 lanes of each half
#pragma unroll
    for (int msk = 1; msk < 16; msk <<= 1)
#pragma unroll
      for (int v = 0; v < 8; ++v) t[v] = fmaxf(t[v], __shfl_xor(t[v], msk, 32));

    float alpha[8];
#pragma unroll
    for (int v = 0; v < 8; ++v) {
      const float nm = fmaxf(mrow[v], t[v]);
      alpha[v] = fast_exp(mrow[v] - nm);
      mrow[v] = nm;
      p0[v] = fast_exp(p0[v] - nm);
      p1[v] = fast_exp(p1[v] - nm);
      t[v] = p0[v] + p1[v];
    }
#pragma unroll
    for (int msk = 1; msk < 16; msk <<= 1)
#pragma unroll
      for (int v = 0; v < 8; ++v) t[v] += __shfl_xor(t[v], msk, 32);
#pragma unroll
    for (int v = 0; v < 8; ++v) {
      ssum[v] = ssum[v] * alpha[v] + t[v];
      acc0[v] *= alpha[v];
      acc1[v] *= alpha[v];
      acc2[v] *= alpha[v];
      acc3[v] *= alpha[v];
    }

    // P tile (16x32) C-layout -> LDS -> A-layout fragment (same-wave, in-order)
#pragma unroll
    for (int v = 0; v < 8; ++v) {
      pb[(8 * hi + v) * 32 + ln] = (bf16_t)p0[v];
      pb[(8 * hi + v) * 32 + 16 + ln] = (bf16_t)p1[v];
    }
    const bf16_t* pr = pb + ln * 32;
    const v16bf ap = load_frag(pr + hi * 16, pr + hi * 16 + 8);

    // V^T B-fragments: B[k][d] = Vt[d, k]; 32B contiguous in k per lane.
    const bf16_t* v0 = Vp + (size_t)(0 + ln) * S_ + kb + hi * 16;
    const bf16_t* v1 = Vp + (size_t)(16 + ln) * S_ + kb + hi * 16;
    const bf16_t* v2 = Vp + (size_t)(32 + ln) * S_ + kb + hi * 16;
    const bf16_t* v3 = Vp + (size_t)(48 + ln) * S_ + kb + hi * 16;
    acc0 = wmma_bf16(ap, load_frag(v0, v0 + 8), acc0);
    acc1 = wmma_bf16(ap, load_frag(v1, v1 + 8), acc1);
    acc2 = wmma_bf16(ap, load_frag(v2, v2 + 8), acc2);
    acc3 = wmma_bf16(ap, load_frag(v3, v3 + 8), acc3);
  }

  // normalize + store: out[b, h*64 + d, s]  (f32)
  float* ob = out + ((size_t)b * EMB + (size_t)h * HDIM) * S_;
#pragma unroll
  for (int v = 0; v < 8; ++v) {
    const float inv = 1.0f / ssum[v];
    const int srow = qbase + v + 8 * hi;
    ob[(size_t)(0 + ln) * S_ + srow] = acc0[v] * inv;
    ob[(size_t)(16 + ln) * S_ + srow] = acc1[v] * inv;
    ob[(size_t)(32 + ln) * S_ + srow] = acc2[v] * inv;
    ob[(size_t)(48 + ln) * S_ + srow] = acc3[v] * inv;
  }
}

// ---------------------------------------------------------------------------
extern "C" void kernel_launch(void* const* d_in, const int* in_sizes, int n_in,
                              void* d_out, int out_size, void* d_ws, size_t ws_size,
                              hipStream_t stream) {
  const float* query = (const float*)d_in[0];
  const float* key = (const float*)d_in[1];
  const float* Wq = (const float*)d_in[2];
  const float* bq = (const float*)d_in[3];
  const float* Wk = (const float*)d_in[4];
  const float* bk = (const float*)d_in[5];
  const float* Wv = (const float*)d_in[6];
  const float* bv = (const float*)d_in[7];
  float* out = (float*)d_out;

  char* ws = (char*)d_ws;
  size_t off = 0;
  bf16_t* Xq = (bf16_t*)(ws + off); off += (size_t)B_ * S_ * CDIM * 2;
  bf16_t* Xk = (bf16_t*)(ws + off); off += (size_t)B_ * S_ * CDIM * 2;
  bf16_t* Wqb = (bf16_t*)(ws + off); off += (size_t)EMB * CDIM * 2;
  bf16_t* Wkb = (bf16_t*)(ws + off); off += (size_t)EMB * CDIM * 2;
  bf16_t* Wvb = (bf16_t*)(ws + off); off += (size_t)EMB * CDIM * 2;
  bf16_t* Qb = (bf16_t*)(ws + off); off += (size_t)B_ * S_ * EMB * 2 / 8 * 8; // (B,H,S,D)
  bf16_t* Kb = (bf16_t*)(ws + off); off += (size_t)B_ * S_ * EMB * 2;
  bf16_t* Vtb = (bf16_t*)(ws + off); off += (size_t)B_ * S_ * EMB * 2;
  (void)ws_size; (void)in_sizes; (void)n_in; (void)out_size;

  {
    const int n = 3 * EMB * CDIM;
    cvt_weights<<<(n + 255) / 256, 256, 0, stream>>>(Wq, Wk, Wv, Wqb, Wkb, Wvb);
  }
  {
    const int n = 2 * B_ * S_ * CDIM;
    cvt_inputs<<<(n + 255) / 256, 256, 0, stream>>>(query, key, Xq, Xk);
  }
  {
    dim3 grid(S_ / 16, EMB / 128, 3 * B_);
    proj_kernel<<<grid, 256, 0, stream>>>(Xq, Xk, Wqb, Wkb, Wvb, bq, bk, bv,
                                          Qb, Kb, Vtb);
  }
  {
    dim3 grid(S_ / 128, B_ * HEADS_);
    attn_kernel<<<grid, 256, 0, stream>>>(Qb, Kb, Vtb, out);
  }
}
